// LiquidEmbedding_8555574854063
// MI455X (gfx1250) — matrix operand order repaired and checked by reading, compiled, not run on hardware
//
#include <hip/hip_runtime.h>
#include <hip/hip_bf16.h>

// ---------------------------------------------------------------------------
// LiquidEmbedding for MI455X (gfx1250, wave32, WMMA + TDM)
//   B=16, S=2048, E=512
// embed(bf16) -> conv1(WMMA,K=3,relu) -> conv2(WMMA,relu,f32)
//   -> FFT per (b,e) (LDS radix-2) + complexity -> N[b] -> masked iFFT
//   -> transpose -> proj GEMM (WMMA) -> d_out ; loss == 0 exactly.
// GEMMs: LDS double-buffered tiles fed by the Tensor Data Mover:
//   one 2-D TDM descriptor for the A tile (130x32, conv halo rows),
//   one 3-D TDM descriptor for all KK B weight tiles (KK x 128 x 32),
//   s_wait_tensorcnt + barrier; WMMA fragments via ds_load_b128.
// Halo rows zeroed in LDS by wave 0 after the tensor wait (block-uniform),
// so the WMMA loop has no per-fragment masking.
// ---------------------------------------------------------------------------

#define B_ 16
#define S_ 2048
#define E_ 512
#define BS_ (B_ * S_)

typedef __attribute__((ext_vector_type(16))) __bf16   v16bf;
typedef __attribute__((ext_vector_type(8)))  float    v8f;
typedef __attribute__((ext_vector_type(4)))  unsigned u32x4;
typedef __attribute__((ext_vector_type(8)))  int      i32x8;
typedef __attribute__((ext_vector_type(4)))  int      i32x4;

union ABfrag { u32x4 u[2]; v16bf v; };

#if defined(__has_builtin)
#  if __has_builtin(__builtin_amdgcn_tensor_load_to_lds)
#    define HAVE_TDM 1
#  else
#    define HAVE_TDM 0
#  endif
#  if __has_builtin(__builtin_amdgcn_s_wait_tensorcnt)
#    define HAVE_TWAIT 1
#  else
#    define HAVE_TWAIT 0
#  endif
#else
#  define HAVE_TDM 0
#  define HAVE_TWAIT 0
#endif

// generic -> LDS(as3) byte address for TDM descriptors
__device__ inline unsigned lds_addr_of(const void* p)
{
    return (unsigned)(unsigned long long)
        (__attribute__((address_space(3))) const void*)p;
}

__device__ inline void wait_tensor0()
{
#if HAVE_TWAIT
    __builtin_amdgcn_s_wait_tensorcnt(0);
#else
    asm volatile("s_wait_tensorcnt 0x0" ::: "memory");
#endif
}

#if HAVE_TDM
// Tile load: [nz] x rows x cols (bf16 elems); row stride stride0 elems,
// plane stride stride2 elems. nz<=1 -> plain 2-D tile (tile_dim2 = 0).
// D# per CDNA5 ISA.
__device__ inline void tdm_load_tiles(const void* gptr, unsigned lds_byte_addr,
                                      int rows, int cols, int stride0,
                                      int nz, long long stride2)
{
    unsigned long long ga = (unsigned long long)gptr;

    u32x4 g0;
    g0.x = 1u;                                   // count=1, user descriptor
    g0.y = lds_byte_addr;                        // lds_addr (bytes)
    g0.z = (unsigned)(ga & 0xffffffffu);         // global_addr[31:0]
    g0.w = (unsigned)((ga >> 32) & 0x01ffffffu)  // global_addr[56:32]
         | 0x80000000u;                          // type=2 -> bits[127:126]=10b

    unsigned td0 = (unsigned)stride0;            // tensor_dim0 (no OOB)
    unsigned td1 = 0xffffu;                      // tensor_dim1 (plenty)
    unsigned st0 = (unsigned)stride0;            // tensor_dim0_stride
    unsigned long long st1 = (unsigned long long)stride2; // tensor_dim1_stride
    unsigned tz  = (nz > 1) ? (unsigned)nz : 0u; // tile_dim2 (0 = unused)

    i32x8 g1;
    g1[0] = (int)(1u << 16);                     // data_size=1 (2 bytes)
    g1[1] = (int)((td0 & 0xffffu) << 16);        // dim0[15:0] @ bits[63:48]
    g1[2] = (int)(((td0 >> 16) & 0xffffu) | ((td1 & 0xffffu) << 16));
    g1[3] = (int)(((td1 >> 16) & 0xffffu) |
                  (((unsigned)cols & 0xffffu) << 16));   // tile_dim0
    g1[4] = (int)(((unsigned)rows & 0xffffu) | (tz << 16)); // tile_dim1|dim2
    g1[5] = (int)st0;                            // dim0_stride[31:0]
    g1[6] = (int)((st1 & 0xffffu) << 16);        // dim1_stride[15:0]
    g1[7] = (int)((st1 >> 16) & 0xffffffffu);    // dim1_stride[47:16]

    i32x4 g2;
    g2.x = 0x7fffffff;                           // tensor_dim2 (no OOB)
    g2.y = 0;                                    // tensor_dim3 / lds incr
    g2.z = 0;                                    // dim2_stride lo
    g2.w = 0;                                    // dim2_stride hi | tile_dim3

#if __clang_major__ >= 23
    __builtin_amdgcn_tensor_load_to_lds(g0, g1, g2, (i32x4)0, (i32x8)0, 0);
#else
    __builtin_amdgcn_tensor_load_to_lds(g0, g1, g2, (i32x4)0, 0);
#endif
}
#endif // HAVE_TDM

// ---------------------------------------------------------------------------
// 1) token + position embedding -> bf16 [B][S][E]
// ---------------------------------------------------------------------------
__global__ __launch_bounds__(256)
void embed_kernel(const int* __restrict__ tokens,
                  const float* __restrict__ tok_emb,
                  const float* __restrict__ pos_emb,
                  __bf16* __restrict__ xa)
{
    int idx = blockIdx.x * 256 + threadIdx.x;
    int e  = idx & (E_ - 1);
    int bs = idx >> 9;
    int s  = bs & (S_ - 1);
    int tok = tokens[bs];
    float v = tok_emb[(size_t)tok * E_ + e] + pos_emb[(size_t)s * E_ + e];
    xa[idx] = (__bf16)v;
}

// ---------------------------------------------------------------------------
// 2) weight prep
// ---------------------------------------------------------------------------
__global__ __launch_bounds__(256)
void prep_conv_w_kernel(const float* __restrict__ src, __bf16* __restrict__ dst)
{
    int idx = blockIdx.x * 256 + threadIdx.x;
    if (idx >= E_ * E_ * 3) return;
    int k = idx % 3;
    int i = (idx / 3) % E_;
    int o = idx / (3 * E_);
    dst[((size_t)k * E_ + o) * E_ + i] = (__bf16)src[idx];
}

__global__ __launch_bounds__(256)
void prep_proj_w_kernel(const float* __restrict__ src, __bf16* __restrict__ dst)
{
    int idx = blockIdx.x * 256 + threadIdx.x;
    if (idx >= E_ * E_) return;
    dst[idx] = (__bf16)src[idx];
}

// ---------------------------------------------------------------------------
// 3) TDM-staged shifted-GEMM conv / plain GEMM (v_wmma_f32_16x16x32_bf16)
//    Block 256 thr = 8 waves (4Mx2N); block tile 128x128; wave tile 32x64.
//    LDS: Abuf[2][130][32] (halo rows), Bbuf[2][3][128][32], double buffered.
// ---------------------------------------------------------------------------
__global__ __launch_bounds__(256)
void gemm_conv_kernel(const __bf16* __restrict__ X,     // [BS][E]
                      const __bf16* __restrict__ W,     // [KK][E][E]
                      const float*  __restrict__ bias,  // [E]
                      float*  __restrict__ outF,
                      __bf16* __restrict__ outH,
                      int KK, int relu)
{
    __shared__ __align__(16) __bf16 Abuf[2][130][32];
    __shared__ __align__(16) __bf16 Bbuf[2][3][128][32];

    const int lane  = threadIdx.x & 31;
    const int wave  = threadIdx.x >> 5;
    const int waveM = wave & 3;
    const int waveN = wave >> 2;
    const int sel   = lane >> 4;
    const int l16   = lane & 15;
    const int m0    = blockIdx.x * 128;
    const int n0    = blockIdx.y * 128;
    const int s0    = m0 & (S_ - 1);
    const int half  = KK >> 1;
    const int zlo   = (s0 == 0);            // row 0 halo invalid (block-uniform)
    const int zhi   = (s0 == S_ - 128);     // row 129 halo invalid

    v8f acc[2][4];
#pragma unroll
    for (int mi = 0; mi < 2; ++mi)
#pragma unroll
        for (int ni = 0; ni < 4; ++ni)
            acc[mi][ni] = (v8f)0.0f;

#if HAVE_TDM
    // ---- fill buffer 0 ----
    if (wave == 0) {
        tdm_load_tiles(X + ((long long)(m0 - 1) * E_ + 0),
                       lds_addr_of(&Abuf[0][0][0]), 130, 32, E_, 0, 0);
        tdm_load_tiles(W + ((size_t)n0 * E_ + 0),
                       lds_addr_of(&Bbuf[0][0][0][0]), 128, 32, E_,
                       KK, (long long)E_ * E_);
        wait_tensor0();
        // ordered after tensor wait: zero invalid halo rows in LDS
        if (zlo && lane < 4)  *(u32x4*)&Abuf[0][0][lane * 8] = (u32x4)0u;
        if (zhi && lane >= 4 && lane < 8)
            *(u32x4*)&Abuf[0][129][(lane - 4) * 8] = (u32x4)0u;
    }
    __syncthreads();
#else
    // ---- synchronous fallback staging of buffer 0 ----
    {
        for (int c = threadIdx.x; c < 130 * 4; c += 256) {
            int r = c >> 2, p4 = c & 3;
            u32x4 v = (u32x4)0u;
            if (!((zlo && r == 0) || (zhi && r == 129)))
                v = *(const u32x4*)(X + ((long long)(m0 - 1 + r) * E_ + p4 * 8));
            *(u32x4*)&Abuf[0][r][p4 * 8] = v;
        }
        for (int c = threadIdx.x; c < KK * 128 * 4; c += 256) {
            int kk = c >> 9, rr = (c >> 2) & 127, p4 = c & 3;
            const __bf16* g = W + ((size_t)(kk * E_ + n0 + rr) * E_ + p4 * 8);
            *(u32x4*)&Bbuf[0][kk][rr][p4 * 8] = *(const u32x4*)g;
        }
        __syncthreads();
    }
#endif

    int cur = 0;
    for (int i0 = 0; i0 < E_; i0 += 32) {
        const int more = (i0 + 32) < E_;

#if HAVE_TDM
        if (more && wave == 0) {
            int nxt = cur ^ 1, i1 = i0 + 32;
            tdm_load_tiles(X + ((long long)(m0 - 1) * E_ + i1),
                           lds_addr_of(&Abuf[nxt][0][0]), 130, 32, E_, 0, 0);
            tdm_load_tiles(W + ((size_t)n0 * E_ + i1),
                           lds_addr_of(&Bbuf[nxt][0][0][0]), 128, 32, E_,
                           KK, (long long)E_ * E_);
        }
#endif

        // ---- compute on buffer `cur` (halo rows pre-zeroed in LDS) ----
        for (int kk = 0; kk < KK; ++kk) {
            const int off = kk - half;

            ABfrag a[2];
#pragma unroll
            for (int mi = 0; mi < 2; ++mi) {
                int lr = waveM * 32 + mi * 16 + l16 + off + 1; // LDS row 0..129
                const __bf16* p = &Abuf[cur][lr][sel * 8];
                a[mi].u[0] = *(const u32x4*)p;                 // K: sel*8 .. +7
                a[mi].u[1] = *(const u32x4*)(p + 16);          // K: 16+sel*8 ..
            }

#pragma unroll
            for (int ni = 0; ni < 4; ++ni) {
                int lc = waveN * 64 + ni * 16 + l16;
                const __bf16* p = &Bbuf[cur][kk][lc][sel * 8];
                ABfrag bf;
                bf.u[0] = *(const u32x4*)p;                    // K: sel*8 .. +7
                bf.u[1] = *(const u32x4*)(p + 16);             // K: 16+sel*8 ..
#pragma unroll
                for (int mi = 0; mi < 2; ++mi) {
                    acc[mi][ni] = __builtin_amdgcn_wmma_f32_16x16x32_bf16(
                        false, a[mi].v, false, bf.v,
                        (short)0, acc[mi][ni], false, false);
                }
            }
        }

#if HAVE_TDM
        if (wave == 0) {
            wait_tensor0();              // only wave 0 has TENSORcnt pending
            if (more) {
                int nxt = cur ^ 1;
                if (zlo && lane < 4)
                    *(u32x4*)&Abuf[nxt][0][lane * 8] = (u32x4)0u;
                if (zhi && lane >= 4 && lane < 8)
                    *(u32x4*)&Abuf[nxt][129][(lane - 4) * 8] = (u32x4)0u;
            }
        }
        __syncthreads();
#else
        __syncthreads();
        if (more) {
            int nxt = cur ^ 1, i1 = i0 + 32;
            for (int c = threadIdx.x; c < 130 * 4; c += 256) {
                int r = c >> 2, p4 = c & 3;
                u32x4 v = (u32x4)0u;
                if (!((zlo && r == 0) || (zhi && r == 129)))
                    v = *(const u32x4*)(X + ((long long)(m0 - 1 + r) * E_ + i1 + p4 * 8));
                *(u32x4*)&Abuf[nxt][r][p4 * 8] = v;
            }
            for (int c = threadIdx.x; c < KK * 128 * 4; c += 256) {
                int kk = c >> 9, rr = (c >> 2) & 127, p4 = c & 3;
                const __bf16* g = W + ((size_t)(kk * E_ + n0 + rr) * E_ + i1 + p4 * 8);
                *(u32x4*)&Bbuf[nxt][kk][rr][p4 * 8] = *(const u32x4*)g;
            }
            __syncthreads();
        }
#endif
        cur ^= 1;
    }

    // epilogue: bias (+relu), D-fragment scatter
#pragma unroll
    for (int mi = 0; mi < 2; ++mi) {
#pragma unroll
        for (int ni = 0; ni < 4; ++ni) {
            int col = n0 + waveN * 64 + ni * 16 + l16;
            float bv = bias[col];
#pragma unroll
            for (int r = 0; r < 8; ++r) {
                int row = m0 + waveM * 32 + mi * 16 + sel * 8 + r;
                float v = acc[mi][ni][r] + bv;
                if (relu) v = fmaxf(v, 0.0f);
                size_t idx = (size_t)row * E_ + col;
                if (outF) outF[idx] = v;
                if (outH) outH[idx] = (__bf16)v;
            }
        }
    }
}

// ---------------------------------------------------------------------------
// 4) forward FFT per (b,e): radix-2 in LDS; spectrum -> [col][S]; complexity
// ---------------------------------------------------------------------------
__global__ __launch_bounds__(256)
void fft_fwd_kernel(const float* __restrict__ h,
                    float* __restrict__ Fre,
                    float* __restrict__ Fim,
                    unsigned int* __restrict__ cnt)
{
    __shared__ float re[S_];
    __shared__ float im[S_];
    __shared__ float redf[256];
    __shared__ unsigned redu[256];

    const int tid = threadIdx.x;
    const int col = blockIdx.x;          // b*E + e
    const int b = col >> 9;
    const int e = col & (E_ - 1);

    for (int t = tid; t < S_; t += 256) {
        re[t] = h[((size_t)b * S_ + t) * E_ + e];
        im[t] = 0.0f;
    }
    __syncthreads();

    for (int t = tid; t < S_; t += 256) {
        int j = __brev(t) >> 21;
        if (j > t) {
            float tr = re[t]; re[t] = re[j]; re[j] = tr;
            float ti = im[t]; im[t] = im[j]; im[j] = ti;
        }
    }
    __syncthreads();

    for (int st = 0; st < 11; ++st) {
        int len = 1 << st;
        for (int t = tid; t < S_ / 2; t += 256) {
            int pos = t & (len - 1);
            int i0 = ((t >> st) << (st + 1)) + pos;
            int i1 = i0 + len;
            float ang = -3.14159265358979f * (float)pos / (float)len;
            float sn, cs;
            __sincosf(ang, &sn, &cs);
            float xr = re[i1], xi = im[i1];
            float tr = xr * cs - xi * sn;
            float ti = xr * sn + xi * cs;
            re[i1] = re[i0] - tr; im[i1] = im[i0] - ti;
            re[i0] += tr;         im[i0] += ti;
        }
        __syncthreads();
    }

    float lmax = 0.0f;
    for (int t = tid; t < S_; t += 256) {
        float r = re[t], i = im[t];
        size_t o = (size_t)col * S_ + t;
        Fre[o] = r;
        Fim[o] = i;
        lmax = fmaxf(lmax, r * r + i * i);
    }
    redf[tid] = lmax;
    __syncthreads();
    for (int s = 128; s > 0; s >>= 1) {
        if (tid < s) redf[tid] = fmaxf(redf[tid], redf[tid + s]);
        __syncthreads();
    }
    float thr2 = 0.01f * redf[0];
    __syncthreads();

    unsigned c = 0;
    for (int t = tid; t < S_; t += 256) {
        float r = re[t], i = im[t];
        c += (r * r + i * i > thr2) ? 1u : 0u;
    }
    redu[tid] = c;
    __syncthreads();
    for (int s = 128; s > 0; s >>= 1) {
        if (tid < s) redu[tid] += redu[tid + s];
        __syncthreads();
    }
    if (tid == 0) atomicAdd(&cnt[b], redu[0]);
}

// ---------------------------------------------------------------------------
// 5) per-batch N + loss(=0) ; counter zeroing
// ---------------------------------------------------------------------------
__global__ void reduce_n_kernel(const unsigned int* __restrict__ cnt,
                                int* __restrict__ Nbuf,
                                float* __restrict__ loss_out)
{
    int b = threadIdx.x;
    if (b < B_) {
        float complexity = (float)cnt[b] / (float)(S_ * E_);
        float ratio = 0.5f * (1.0f - complexity);
        ratio = fminf(fmaxf(ratio, 0.1f), 1.0f);
        int N = (int)(ratio * (float)S_);
        N = N < 1 ? 1 : (N > S_ ? S_ : N);
        Nbuf[b] = N;
    }
    if (threadIdx.x == 0) *loss_out = 0.0f;
}

__global__ void zero_cnt_kernel(unsigned int* __restrict__ cnt)
{
    if (threadIdx.x < B_) cnt[threadIdx.x] = 0u;
}

// ---------------------------------------------------------------------------
// 6) masked inverse FFT; real/S -> Z[col][S]
// ---------------------------------------------------------------------------
__global__ __launch_bounds__(256)
void ifft_kernel(const float* __restrict__ Fre,
                 const float* __restrict__ Fim,
                 const int* __restrict__ Nbuf,
                 float* __restrict__ Z)
{
    __shared__ float re[S_];
    __shared__ float im[S_];

    const int tid = threadIdx.x;
    const int col = blockIdx.x;
    const int b = col >> 9;
    const int N = Nbuf[b];

    for (int t = tid; t < S_; t += 256) {
        if (t < N) {
            size_t o = (size_t)col * S_ + t;
            re[t] = Fre[o];
            im[t] = Fim[o];
        } else {
            re[t] = 0.0f;
            im[t] = 0.0f;
        }
    }
    __syncthreads();

    for (int t = tid; t < S_; t += 256) {
        int j = __brev(t) >> 21;
        if (j > t) {
            float tr = re[t]; re[t] = re[j]; re[j] = tr;
            float ti = im[t]; im[t] = im[j]; im[j] = ti;
        }
    }
    __syncthreads();

    for (int st = 0; st < 11; ++st) {
        int len = 1 << st;
        for (int t = tid; t < S_ / 2; t += 256) {
            int pos = t & (len - 1);
            int i0 = ((t >> st) << (st + 1)) + pos;
            int i1 = i0 + len;
            float ang = 3.14159265358979f * (float)pos / (float)len;
            float sn, cs;
            __sincosf(ang, &sn, &cs);
            float xr = re[i1], xi = im[i1];
            float tr = xr * cs - xi * sn;
            float ti = xr * sn + xi * cs;
            re[i1] = re[i0] - tr; im[i1] = im[i0] - ti;
            re[i0] += tr;         im[i0] += ti;
        }
        __syncthreads();
    }

    const float scale = 1.0f / (float)S_;
    for (int t = tid; t < S_; t += 256)
        Z[(size_t)col * S_ + t] = re[t] * scale;
}

// ---------------------------------------------------------------------------
// 7) transpose Z [B][E][S] f32 -> A [B][S][E] bf16
// ---------------------------------------------------------------------------
__global__ __launch_bounds__(256)
void transpose_kernel(const float* __restrict__ Z, __bf16* __restrict__ A)
{
    __shared__ float tile[32][33];
    const int b = blockIdx.z;
    const int e0 = blockIdx.y * 32;
    const int s0 = blockIdx.x * 32;
    const int c = threadIdx.x & 31;
    const int r = threadIdx.x >> 5;

#pragma unroll
    for (int it = 0; it < 4; ++it) {
        int e = e0 + r + it * 8;
        tile[r + it * 8][c] = Z[((size_t)b * E_ + e) * S_ + s0 + c];
    }
    __syncthreads();
#pragma unroll
    for (int it = 0; it < 4; ++it) {
        int s = s0 + r + it * 8;
        A[((size_t)b * S_ + s) * E_ + e0 + c] = (__bf16)tile[c][r + it * 8];
    }
}

// ---------------------------------------------------------------------------
// launch
// ---------------------------------------------------------------------------
extern "C" void kernel_launch(void* const* d_in, const int* in_sizes, int n_in,
                              void* d_out, int out_size, void* d_ws, size_t ws_size,
                              hipStream_t stream)
{
    const int*   tokens  = (const int*)d_in[0];
    const float* tok_emb = (const float*)d_in[1];
    const float* pos_emb = (const float*)d_in[2];
    const float* c1w     = (const float*)d_in[3];
    const float* c1b     = (const float*)d_in[4];
    const float* c2w     = (const float*)d_in[5];
    const float* c2b     = (const float*)d_in[6];
    const float* pw      = (const float*)d_in[7];
    const float* pb      = (const float*)d_in[8];
    float* out = (float*)d_out;
    float* loss_out = out + (size_t)BS_ * E_;

    char* wsp = (char*)d_ws;
    auto alloc = [&](size_t n) {
        char* p = wsp;
        wsp += (n + 255) & ~(size_t)255;
        return p;
    };
    __bf16* w1b   = (__bf16*)alloc((size_t)3 * E_ * E_ * 2);
    __bf16* w2b   = (__bf16*)alloc((size_t)3 * E_ * E_ * 2);
    __bf16* projb = (__bf16*)alloc((size_t)E_ * E_ * 2);
    __bf16* xa    = (__bf16*)alloc((size_t)BS_ * E_ * 2);   // embed / proj A
    __bf16* y1    = (__bf16*)alloc((size_t)BS_ * E_ * 2);   // conv1 out
    float*  h     = (float*)alloc((size_t)BS_ * E_ * 4);    // conv2 out / Z
    float*  Fre   = (float*)alloc((size_t)BS_ * E_ * 4);
    float*  Fim   = (float*)alloc((size_t)BS_ * E_ * 4);
    unsigned int* cnt = (unsigned int*)alloc(64);
    int* Nbuf         = (int*)alloc(64);

    zero_cnt_kernel<<<1, 32, 0, stream>>>(cnt);

    embed_kernel<<<(BS_ * E_) / 256, 256, 0, stream>>>(tokens, tok_emb, pos_emb, xa);

    prep_conv_w_kernel<<<(E_ * E_ * 3 + 255) / 256, 256, 0, stream>>>(c1w, w1b);
    prep_conv_w_kernel<<<(E_ * E_ * 3 + 255) / 256, 256, 0, stream>>>(c2w, w2b);
    prep_proj_w_kernel<<<(E_ * E_ + 255) / 256, 256, 0, stream>>>(pw, projb);

    dim3 ggrid(BS_ / 128, E_ / 128);
    gemm_conv_kernel<<<ggrid, 256, 0, stream>>>(xa, w1b, c1b, nullptr, y1, 3, 1);
    gemm_conv_kernel<<<ggrid, 256, 0, stream>>>(y1, w2b, c2b, h, nullptr, 3, 1);

    fft_fwd_kernel<<<B_ * E_, 256, 0, stream>>>(h, Fre, Fim, cnt);

    reduce_n_kernel<<<1, 32, 0, stream>>>(cnt, Nbuf, loss_out);

    ifft_kernel<<<B_ * E_, 256, 0, stream>>>(Fre, Fim, Nbuf, h);

    transpose_kernel<<<dim3(S_ / 32, E_ / 32, B_), 256, 0, stream>>>(h, xa);

    gemm_conv_kernel<<<ggrid, 256, 0, stream>>>(xa, projb, pb, out, nullptr, 1, 0);
}
